// GPT2Attention_64269890617349
// MI455X (gfx1250) — matrix operand order repaired and checked by reading, not compiled, not run on hardware
//
#include <hip/hip_runtime.h>

typedef unsigned short u16;
typedef __attribute__((ext_vector_type(16))) __bf16 bf16x16;
typedef __attribute__((ext_vector_type(8)))  float   f32x8;

union FragU {
  bf16x16 bf;
  uint4   q[2];
  u16     u[16];
};
union AccU {
  f32x8 v;
  float f[8];
};

__device__ __forceinline__ u16 f2bf(float x) {
  union { float f; unsigned u; } v; v.f = x;
  unsigned r = v.u + 0x7FFFu + ((v.u >> 16) & 1u);  // round-to-nearest-even
  return (u16)(r >> 16);
}

__device__ __forceinline__ f32x8 wmma_bf16(bf16x16 a, bf16x16 b, f32x8 c) {
  // (neg_a, A, neg_b, B, c_mod, C, reuse_a, reuse_b)
  return __builtin_amdgcn_wmma_f32_16x16x32_bf16(false, a, false, b, (short)0, c,
                                                 false, false);
}

// Async DMA: global (16B per lane) -> LDS, tracked by ASYNCcnt (no VGPR data).
__device__ __forceinline__ void async_copy16(u16* lds_dst, const u16* gsrc) {
  const unsigned laddr = (unsigned)(unsigned long long)lds_dst;  // LDS aperture: low 32 bits
  asm volatile("global_load_async_to_lds_b128 %0, %1, off"
               :
               : "v"(laddr), "v"(gsrc)
               : "memory");
}
__device__ __forceinline__ void wait_async0() {
  asm volatile("s_wait_asynccnt 0x0" ::: "memory");
}

// A fragment: 16 (rows) x 32 (K) bf16 from row-major [row][K] source.
// lanes 0-15: M=lane, K = {0..7, 16..23}; lanes 16-31: M=lane-16, K = {8..15, 24..31}
__device__ __forceinline__ bf16x16 load_fragA(const u16* base, int stride, int lane) {
  const int half = lane >> 4;
  const int r    = lane & 15;
  const u16* p = base + r * stride + half * 8;
  FragU f;
  f.q[0] = *(const uint4*)(p);
  f.q[1] = *(const uint4*)(p + 16);
  return f.bf;
}

// B fragment: 32 (K) x 16 (N) bf16 from [N][K] row-major source.
// lanes 0-15: N=lane, K = {0..7, 16..23}; lanes 16-31: N=lane-16, K = {8..15, 24..31}
__device__ __forceinline__ bf16x16 load_fragB(const u16* base, int stride, int lane) {
  const int half = lane >> 4;
  const int n    = lane & 15;
  const u16* p = base + n * stride + half * 8;
  FragU f;
  f.q[0] = *(const uint4*)(p);
  f.q[1] = *(const uint4*)(p + 16);
  return f.bf;
}

// ---------------------------------------------------------------- converts
__global__ void cvt_f32_bf16(const float* __restrict__ src, u16* __restrict__ dst,
                             int n) {
  int i = blockIdx.x * blockDim.x + threadIdx.x;
  const int stride = gridDim.x * blockDim.x;
  for (; i < n; i += stride) dst[i] = f2bf(src[i]);
}

// W [1024][3072] fp32 -> Wt [3072][1024] bf16 (one-off 6MB transform; makes the
// GEMM B-tile a straight row-copy so it can ride the async-DMA path).
__global__ void cvt_transpose(const float* __restrict__ W, u16* __restrict__ Wt) {
  const int idx = blockIdx.x * blockDim.x + threadIdx.x;
  if (idx < 1024 * 3072) {
    const int k = idx >> 11 == 0 ? 0 : 0, kk = idx / 3072, n = idx % 3072;
    (void)k;
    Wt[(size_t)n * 1024 + kk] = f2bf(W[idx]);
  }
}

// ---------------------------------------------------------------- QKV GEMM
// C[8192,3072] = A[8192,1024] x Wt^T + bias, scattered to head-split
// bf16 Q/K/V [b=4][h=16][s=2048][c=64]. Double-buffered async-LDS pipeline.
constexpr int AS = 40;  // 32 K + 8 pad (conflict-free b128 frag loads)

// copy [128 rows][32 cols] (row stride 1024) into LDS tile (stride AS), async
__device__ __forceinline__ void async_tile128x32(const u16* __restrict__ g,
                                                 u16* lbuf, int tid) {
  for (int i = tid; i < 512; i += 256) {
    const int row = i >> 2, kc = i & 3;
    async_copy16(lbuf + row * AS + kc * 8, g + (size_t)row * 1024 + kc * 8);
  }
}

__global__ __launch_bounds__(256)
void qkv_gemm(const u16* __restrict__ Abf, const u16* __restrict__ Wt,
              const float* __restrict__ bias,
              u16* __restrict__ Qg, u16* __restrict__ Kg, u16* __restrict__ Vg) {
  __shared__ alignas(16) u16 As[2][128 * AS];
  __shared__ alignas(16) u16 Bs[2][128 * AS];

  const int tid   = threadIdx.x;
  const int lane  = tid & 31;
  const int wid   = tid >> 5;
  const int waveM = wid & 1;   // 2 wave-rows of 64
  const int waveN = wid >> 1;  // 4 wave-cols of 32
  const int m0 = blockIdx.y * 128;
  const int n0 = blockIdx.x * 128;
  const int half = lane >> 4;
  const int col  = lane & 15;

  const u16* Ag = Abf + (size_t)m0 * 1024;  // + k0
  const u16* Bg = Wt + (size_t)n0 * 1024;   // + k0

  AccU acc[4][2];
  for (int mf = 0; mf < 4; ++mf)
    for (int nf = 0; nf < 2; ++nf)
      for (int r = 0; r < 8; ++r) acc[mf][nf].f[r] = 0.0f;

  float biasv[2];
  for (int nf = 0; nf < 2; ++nf)
    biasv[nf] = bias[n0 + waveN * 32 + nf * 16 + col];

  // prime the pipeline: tile 0 -> buffer 0
  async_tile128x32(Ag + 0, As[0], tid);
  async_tile128x32(Bg + 0, Bs[0], tid);

  for (int it = 0; it < 32; ++it) {
    wait_async0();       // this wave's async copies into buf[it&1] complete
    __syncthreads();     // => all waves' copies complete; prior buf reads done

    const int cur = it & 1;
    if (it + 1 < 32) {   // prefetch next tile into the other buffer
      async_tile128x32(Ag + (it + 1) * 32, As[cur ^ 1], tid);
      async_tile128x32(Bg + (it + 1) * 32, Bs[cur ^ 1], tid);
    }

    const u16* Aw = As[cur] + (waveM * 64) * AS;
    const u16* Bw = Bs[cur] + (waveN * 32) * AS;
    bf16x16 af[4], bfr[2];
    for (int mf = 0; mf < 4; ++mf) af[mf] = load_fragA(Aw + mf * 16 * AS, AS, lane);
    for (int nf = 0; nf < 2; ++nf) bfr[nf] = load_fragB(Bw + nf * 16 * AS, AS, lane);
    for (int mf = 0; mf < 4; ++mf)
      for (int nf = 0; nf < 2; ++nf)
        acc[mf][nf].v = wmma_bf16(af[mf], bfr[nf], acc[mf][nf].v);
  }

  // bias + scatter: n<1024 -> Q, <2048 -> K, else V; [b][h][s][c] bf16
  u16* dsts[3] = {Qg, Kg, Vg};
  for (int mf = 0; mf < 4; ++mf) {
    for (int nf = 0; nf < 2; ++nf) {
      const int n     = n0 + waveN * 32 + nf * 16 + col;
      const int which = n >> 10;
      const int hh    = n & 1023;
      const int head  = hh >> 6;
      const int c     = hh & 63;
      u16* dst = dsts[which];
      for (int r = 0; r < 8; ++r) {
        const int m    = m0 + waveM * 64 + mf * 16 + r + 8 * half;
        const int b    = m >> 11;
        const int srow = m & 2047;
        dst[(((size_t)b * 16 + head) * 2048 + srow) * 64 + c] =
            f2bf(acc[mf][nf].f[r] + biasv[nf]);
      }
    }
  }
}

// ---------------------------------------------------------------- attention
// Flash-attention: block = (batch*head, 64-query tile), 4 waves x 16 rows.
constexpr int KS = 72;  // 64 + 8 pad
constexpr int VS = 72;
constexpr int PS = 72;

__global__ __launch_bounds__(128)
void attn(const u16* __restrict__ Qg, const u16* __restrict__ Kg,
          const u16* __restrict__ Vg, float* __restrict__ out) {
  __shared__ alignas(16) u16 Ks[64 * KS];  // [key][c]
  __shared__ alignas(16) u16 Vt[64 * VS];  // [c][key] (transposed)
  __shared__ alignas(16) u16 Ps[64 * PS];  // [q][key] bf16 P, per-wave 16-row slab

  const int tid  = threadIdx.x;
  const int lane = tid & 31;
  const int w    = tid >> 5;
  const int half = lane >> 4;
  const int col  = lane & 15;
  const int qt = blockIdx.x;  // 0..31 query tile
  const int bh = blockIdx.y;  // 0..63
  const int b = bh >> 4, h = bh & 15;

  const u16* Qb = Qg + (size_t)bh * 2048 * 64;
  const u16* Kb = Kg + (size_t)bh * 2048 * 64;
  const u16* Vb = Vg + (size_t)bh * 2048 * 64;

  const int qrow0 = qt * 64 + w * 16;

  // Q fragments live in registers for the whole key loop
  bf16x16 qf[2];
  qf[0] = load_fragA(Qb + (size_t)qrow0 * 64 + 0, 64, lane);
  qf[1] = load_fragA(Qb + (size_t)qrow0 * 64 + 32, 64, lane);

  AccU o[4];
  float mrow[8], lrow[8];
  for (int cf = 0; cf < 4; ++cf)
    for (int r = 0; r < 8; ++r) o[cf].f[r] = 0.0f;
  for (int r = 0; r < 8; ++r) { mrow[r] = -1e30f; lrow[r] = 0.0f; }

  for (int kt = 0; kt <= qt; ++kt) {  // causal: skip tiles fully above diagonal
    __syncthreads();                  // prior reads of Ks/Vt done before refill
    {
      // K tile: async DMA straight into LDS (4 x b128 per thread)
      for (int i = tid; i < 512; i += 128) {
        const int key = i >> 2, c8 = (i & 3) * 8;
        async_copy16(Ks + key * KS + c8,
                     Kb + (size_t)(kt * 64 + key) * 64 + c8);
      }
      // V tile: needs transpose -> manual path
      const int key = tid >> 1;
      const int ch  = (tid & 1) * 32;
      const uint4* v4 = (const uint4*)(Vb + (size_t)(kt * 64 + key) * 64 + ch);
      FragU t0, t1;
      t0.q[0] = v4[0]; t0.q[1] = v4[1];
      t1.q[0] = v4[2]; t1.q[1] = v4[3];
      for (int j = 0; j < 16; ++j) Vt[(ch + j) * VS + key]      = t0.u[j];
      for (int j = 0; j < 16; ++j) Vt[(ch + 16 + j) * VS + key] = t1.u[j];
    }
    wait_async0();
    __syncthreads();

    // S = Q K^T  (16q x 64k per wave), K dim = c = 64 -> 2 wmma steps
    AccU s[4];
    for (int nf = 0; nf < 4; ++nf) {
      AccU z;
      for (int r = 0; r < 8; ++r) z.f[r] = 0.0f;
      z.v = wmma_bf16(qf[0], load_fragB(Ks + nf * 16 * KS + 0, KS, lane), z.v);
      z.v = wmma_bf16(qf[1], load_fragB(Ks + nf * 16 * KS + 32, KS, lane), z.v);
      s[nf] = z;
    }

    // scale + causal mask (diagonal tile only)
    const float rsc = 0.125f;  // 1/sqrt(64)
    if (kt == qt) {
      for (int nf = 0; nf < 4; ++nf)
        for (int r = 0; r < 8; ++r) {
          const int qg = qrow0 + r + 8 * half;
          const int kg = kt * 64 + nf * 16 + col;
          const float v = s[nf].f[r] * rsc;
          s[nf].f[r] = (kg <= qg) ? v : -10000.0f;
        }
    } else {
      for (int nf = 0; nf < 4; ++nf)
        for (int r = 0; r < 8; ++r) s[nf].f[r] *= rsc;
    }

    // online softmax; each half-wave's lanes share rows r(+8*half),
    // reduce across the 16 lanes of a half with width-16 xor shuffles
    float mnew[8], alpha[8];
    for (int r = 0; r < 8; ++r) {
      float mx = fmaxf(fmaxf(s[0].f[r], s[1].f[r]), fmaxf(s[2].f[r], s[3].f[r]));
      for (int off = 1; off < 16; off <<= 1)
        mx = fmaxf(mx, __shfl_xor(mx, off, 16));
      mnew[r]  = fmaxf(mrow[r], mx);
      alpha[r] = __expf(mrow[r] - mnew[r]);
      mrow[r]  = mnew[r];
    }
    for (int nf = 0; nf < 4; ++nf)
      for (int r = 0; r < 8; ++r) s[nf].f[r] = __expf(s[nf].f[r] - mnew[r]);
    for (int r = 0; r < 8; ++r) {
      float sum = s[0].f[r] + s[1].f[r] + s[2].f[r] + s[3].f[r];
      for (int off = 1; off < 16; off <<= 1)
        sum += __shfl_xor(sum, off, 16);
      lrow[r] = lrow[r] * alpha[r] + sum;
    }
    for (int cf = 0; cf < 4; ++cf)
      for (int r = 0; r < 8; ++r) o[cf].f[r] *= alpha[r];

    // C-layout -> A-layout for P via per-wave LDS slab (no cross-wave sync)
    u16* Pw = Ps + (w * 16) * PS;
    for (int nf = 0; nf < 4; ++nf)
      for (int r = 0; r < 8; ++r)
        Pw[(r + 8 * half) * PS + nf * 16 + col] = f2bf(s[nf].f[r]);

    // O += P V, K dim = key = 64 -> 2 wmma steps per c-frag
    for (int cf = 0; cf < 4; ++cf) {
      f32x8 accv = o[cf].v;
      accv = wmma_bf16(load_fragA(Pw + 0, PS, lane),
                       load_fragB(Vt + cf * 16 * VS + 0, VS, lane), accv);
      accv = wmma_bf16(load_fragA(Pw + 32, PS, lane),
                       load_fragB(Vt + cf * 16 * VS + 32, VS, lane), accv);
      o[cf].v = accv;
    }
  }

  // normalize and write fp32 output [b][s][hidden]
  for (int r = 0; r < 8; ++r) {
    const float inv = 1.0f / lrow[r];
    const int q = qrow0 + r + 8 * half;
    float* orow = out + ((size_t)b * 2048 + q) * 1024 + h * 64;
    for (int cf = 0; cf < 4; ++cf)
      orow[cf * 16 + col] = o[cf].f[r] * inv;
  }
}

// ---------------------------------------------------------------- launcher
extern "C" void kernel_launch(void* const* d_in, const int* in_sizes, int n_in,
                              void* d_out, int out_size, void* d_ws, size_t ws_size,
                              hipStream_t stream) {
  const float* enc  = (const float*)d_in[0];  // [4,2048,1024]
  const float* W    = (const float*)d_in[1];  // [1024,3072]
  const float* bias = (const float*)d_in[2];  // [3072]
  float* out = (float*)d_out;

  u16* ws  = (u16*)d_ws;
  u16* Abf = ws;                               // 8192*1024       (16 MB)
  u16* Wt  = Abf + (size_t)8192 * 1024;        // 3072*1024 (T)   ( 6 MB)
  u16* Qg  = Wt + (size_t)3072 * 1024;         // 64*2048*64      (16 MB)
  u16* Kg  = Qg + (size_t)64 * 2048 * 64;      //                 (16 MB)
  u16* Vg  = Kg + (size_t)64 * 2048 * 64;      //                 (16 MB)

  cvt_f32_bf16<<<2048, 256, 0, stream>>>(enc, Abf, 8192 * 1024);
  cvt_transpose<<<(1024 * 3072 + 255) / 256, 256, 0, stream>>>(W, Wt);
  qkv_gemm<<<dim3(24, 64), 256, 0, stream>>>(Abf, Wt, bias, Qg, Kg, Vg);
  attn<<<dim3(32, 64), 128, 0, stream>>>(Qg, Kg, Vg, out);
}
